// WindowAttention_34634616275470
// MI455X (gfx1250) — matrix operand order repaired and checked by reading, not compiled, not run on hardware
//
#include <hip/hip_runtime.h>
#include <hip/hip_bf16.h>

typedef _Float16 h16;
typedef __attribute__((ext_vector_type(16))) _Float16 v16h;
typedef __attribute__((ext_vector_type(8)))  _Float16 v8h;
typedef __attribute__((ext_vector_type(4)))  _Float16 v4h;
typedef __attribute__((ext_vector_type(8)))  float    v8f;
typedef __attribute__((ext_vector_type(4)))  unsigned int u32x4;
typedef __attribute__((ext_vector_type(8)))  int          i32x8;
typedef __attribute__((ext_vector_type(4)))  int          i32x4;

#define DIMC   256
#define HEADS  8
#define NTOK   49
#define NPAD   64
#define DH     32
#define NWIN   64

// ---- LDS layout, BYTE offsets (all fragment rows 16B aligned) ----
#define XS 264          /* stride in halves for xA/q/k/oA rows            */
#define VS 72           /* vT row stride (halves)                          */
#define PS 64           /* p   row stride (halves)                         */
#define OFF_X   0       /* xA  f16 64x264  -> 33792 B                      */
#define OFF_Q   33792   /* q   f16 64x264  -> 33792 B                      */
#define OFF_K   67584   /* k   f16 64x264  -> 33792 B                      */
#define OFF_V   101376  /* vT  f16 256x72  -> 36864 B, total 138240        */
#define OFF_XS  33792   /* f32 x staging (50176 B) overlays q+k (pre-GEMM) */
#define OFF_P   33792   /* p   f16 8x64x64 (65536 B) overlays dead q,k     */
#define OFF_O   0       /* oA  f16 64x264 overlays dead xA                 */
#define LDS_BYTES 138240

// ---- workspace layout (bytes) ----
#define WS_WT  0         // qkv_w^T  f16 [768][256]           (393216 B)
#define WS_PT  393216    // proj_w^T f16 [256][256]           (131072 B)
#define WS_BM  524288    // fused bias+mask f32 [64][8][49][49] (4917248 B)

__device__ __forceinline__ v8f wmma16(v16h a, v16h b, v8f c) {
    return __builtin_amdgcn_wmma_f32_16x16x32_f16(false, a, false, b,
                                                  (short)0, c, false, false);
}

// A fragment 16x32 f16 from row-major [row][k], stride S halves.
__device__ __forceinline__ v16h frag_A(const h16* p, int S, int lane) {
    int r  = lane & 15;
    int kb = (lane >> 4) << 3;
    const h16* q = p + r * S + kb;
    v8h lo = *(const v8h*)(q);
    v8h hi = *(const v8h*)(q + 16);
    return __builtin_shufflevector(lo, hi, 0,1,2,3,4,5,6,7,8,9,10,11,12,13,14,15);
}

// B fragment 32x16 f16 from N-major (transposed) storage Bt[n][k], stride S halves.
__device__ __forceinline__ v16h frag_B(const h16* p, int S, int lane) {
    int n  = lane & 15;
    int kb = (lane >> 4) << 3;
    const h16* q = p + n * S + kb;
    v8h lo = *(const v8h*)(q);
    v8h hi = *(const v8h*)(q + 16);
    return __builtin_shufflevector(lo, hi, 0,1,2,3,4,5,6,7,8,9,10,11,12,13,14,15);
}

__global__ void swin_prep(const float* __restrict__ qkv_w,      // [256][768]
                          const float* __restrict__ proj_w,     // [256][256]
                          const float* __restrict__ bias_table, // [169][8]
                          const int*   __restrict__ rel_index,  // [49*49]
                          const float* __restrict__ mask,       // [64][49][49]
                          h16* __restrict__ wt, h16* __restrict__ pt,
                          float* __restrict__ bm) {
    int stride = gridDim.x * blockDim.x;
    for (int i = blockIdx.x * blockDim.x + threadIdx.x; i < 768 * 256; i += stride) {
        int j = i >> 8, k = i & 255;                 // wt[j][k] = qkv_w[k][j]
        wt[i] = (h16)qkv_w[k * 768 + j];
    }
    for (int i = blockIdx.x * blockDim.x + threadIdx.x; i < 256 * 256; i += stride) {
        int j = i >> 8, k = i & 255;
        pt[i] = (h16)proj_w[k * 256 + j];
    }
    // fused bm[w][h][n][m] = bias_table[rel_index[n*49+m]][h] + mask[w][n][m]
    for (int i = blockIdx.x * blockDim.x + threadIdx.x;
         i < NWIN * HEADS * NTOK * NTOK; i += stride) {
        int nm = i % (NTOK * NTOK);
        int wh = i / (NTOK * NTOK);
        int h  = wh & (HEADS - 1);
        int w  = wh >> 3;
        bm[i] = bias_table[rel_index[nm] * HEADS + h] + mask[w * NTOK * NTOK + nm];
    }
}

__global__ __launch_bounds__(256) void swin_attn(
        const float* __restrict__ x,        // [4096][49][256]
        const float* __restrict__ qkv_b,    // [768]
        const float* __restrict__ proj_b,   // [256]
        const h16*   __restrict__ wt,       // [768][256] f16 (N-major)
        const h16*   __restrict__ pt,       // [256][256] f16 (N-major)
        const float* __restrict__ bm,       // [64][8][49][49]
        float* __restrict__ out) {          // [4096][49][256]
    extern __shared__ char smem[];
    h16*   xA  = (h16*)(smem + OFF_X);
    h16*   qls = (h16*)(smem + OFF_Q);
    h16*   kls = (h16*)(smem + OFF_K);
    h16*   vls = (h16*)(smem + OFF_V);
    h16*   pls = (h16*)(smem + OFF_P);
    h16*   oA  = (h16*)(smem + OFF_O);
    float* xst = (float*)(smem + OFF_XS);

    const int tid  = threadIdx.x;
    const int wave = tid >> 5;
    const int lane = tid & 31;
    const int bid  = blockIdx.x;
    const int mid  = bid & (NWIN - 1);
    const float scale = 0.17677669529663687f;   // 1/sqrt(32)

    // ---- TDM: DMA x window (49x256 f32, 50176 B) into LDS staging ----
    if (wave == 0) {
        unsigned long long ga =
            (unsigned long long)(const void*)(x + (size_t)bid * NTOK * DIMC);
        u32x4 g0;
        g0[0] = 1u;                                          // count=1, user D#
        g0[1] = (unsigned)(size_t)(smem + OFF_XS);           // lds_addr (bytes)
        g0[2] = (unsigned)(ga & 0xFFFFFFFFu);                // global_addr[31:0]
        g0[3] = (unsigned)((ga >> 32) & 0x01FFFFFFu) | 0x80000000u; // [56:32]|type=2
        i32x8 g1;
        g1[0] = 0x00020000;          // data_size = 2 -> 4-byte elements
        g1[1] = (int)(256u << 16);   // tensor_dim0 = 256
        g1[2] = (int)(49u  << 16);   // tensor_dim1 = 49
        g1[3] = (int)(256u << 16);   // tile_dim0   = 256
        g1[4] = 49;                  // tile_dim1   = 49, tile_dim2 = 0
        g1[5] = 256;                 // tensor_dim0_stride = 256
        g1[6] = 0;
        g1[7] = 0;
        i32x4 g2 = (i32x4)0;
        i32x4 g3 = (i32x4)0;
#if defined(__clang_major__) && (__clang_major__ >= 23)
        i32x8 z8 = (i32x8)0;
        __builtin_amdgcn_tensor_load_to_lds(g0, g1, g2, g3, z8, 0);
#else
        __builtin_amdgcn_tensor_load_to_lds(g0, g1, g2, g3, 0);
#endif
        __builtin_amdgcn_s_wait_tensorcnt(0);
    }
    __syncthreads();

    // ---- convert staged f32 -> f16 xA, pad rows 49..63 with zeros ----
    for (int i = tid; i < NPAD * (DIMC / 4); i += 256) {
        int r = i >> 6, c4 = i & 63;
        v4h hv;
        if (r < NTOK) {
            float4 f = *(const float4*)(xst + r * DIMC + c4 * 4);
            hv[0] = (h16)f.x; hv[1] = (h16)f.y; hv[2] = (h16)f.z; hv[3] = (h16)f.w;
        } else {
            hv[0] = (h16)0.f; hv[1] = (h16)0.f; hv[2] = (h16)0.f; hv[3] = (h16)0.f;
        }
        *(v4h*)(xA + r * XS + c4 * 4) = hv;
    }
    __syncthreads();

    const int col_l = lane & 15;
    const int rofs  = (lane >> 4) << 3;

    // ---- stage 1: qkv = x @ W + b ; 4 x 48 tiles of 16x16, K=256 ----
    for (int t = wave; t < 192; t += 8) {
        int mt = t & 3, nt = t >> 2;
        if (t + 8 < 192)    // prefetch next B panel into GL2/L0
            __builtin_prefetch(wt + ((t + 8) >> 2) * 16 * 256, 0, 3);
        v8f c = 0.0f;
#pragma unroll
        for (int kk = 0; kk < 8; ++kk) {
            v16h a = frag_A(xA + mt * 16 * XS + kk * 32, XS, lane);
            v16h b = frag_B(wt + (nt * 16) * 256 + kk * 32, 256, lane);
            c = wmma16(a, b, c);
        }
        int j0 = nt * 16;
        float bc = qkv_b[j0 + col_l];
        if (j0 < 256) {                 // q (pre-scaled)
#pragma unroll
            for (int r = 0; r < 8; ++r)
                qls[(mt * 16 + rofs + r) * XS + (j0 + col_l)] = (h16)((c[r] + bc) * scale);
        } else if (j0 < 512) {          // k
#pragma unroll
            for (int r = 0; r < 8; ++r)
                kls[(mt * 16 + rofs + r) * XS + (j0 - 256 + col_l)] = (h16)(c[r] + bc);
        } else {                        // v, stored transposed: vT[d][key]
#pragma unroll
            for (int r = 0; r < 8; ++r)
                vls[(j0 - 512 + col_l) * VS + (mt * 16 + rofs + r)] = (h16)(c[r] + bc);
        }
    }
    __syncthreads();

    // ---- stage 2: per-head attn = q @ k^T (wave == head), 16 tiles ----
    const int h = wave;
    v8f acc[4][4];
#pragma unroll
    for (int mt = 0; mt < 4; ++mt) {
        v16h a = frag_A(qls + mt * 16 * XS + h * DH, XS, lane);
#pragma unroll
        for (int nt = 0; nt < 4; ++nt) {
            v16h b = frag_B(kls + (nt * 16) * XS + h * DH, XS, lane);
            v8f z = 0.0f;
            acc[mt][nt] = wmma16(a, b, z);
        }
    }
    __syncthreads();   // q,k dead -> p may overlay them

    // ---- fused bias+mask, softmax over keys, write p (f16) to LDS ----
    {
        const float* bmh = bm + ((size_t)(mid * HEADS + h)) * NTOK * NTOK;
        h16* ph = pls + h * NPAD * PS;
#pragma unroll
        for (int mt = 0; mt < 4; ++mt) {
#pragma unroll
            for (int r = 0; r < 8; ++r) {
                int row = mt * 16 + rofs + r;
                float v0[4];
#pragma unroll
                for (int nt = 0; nt < 4; ++nt) {
                    int cg = nt * 16 + col_l;
                    float tvl = acc[mt][nt][r];
                    if (row < NTOK && cg < NTOK)
                        tvl += bmh[row * NTOK + cg];
                    else
                        tvl = -1e30f;
                    v0[nt] = tvl;
                }
                float mx = fmaxf(fmaxf(v0[0], v0[1]), fmaxf(v0[2], v0[3]));
                mx = fmaxf(mx, __shfl_xor(mx, 1, 32));
                mx = fmaxf(mx, __shfl_xor(mx, 2, 32));
                mx = fmaxf(mx, __shfl_xor(mx, 4, 32));
                mx = fmaxf(mx, __shfl_xor(mx, 8, 32));
                float s = 0.f;
#pragma unroll
                for (int nt = 0; nt < 4; ++nt) { v0[nt] = __expf(v0[nt] - mx); s += v0[nt]; }
                s += __shfl_xor(s, 1, 32);
                s += __shfl_xor(s, 2, 32);
                s += __shfl_xor(s, 4, 32);
                s += __shfl_xor(s, 8, 32);
                float inv = 1.0f / fmaxf(s, 1e-30f);
#pragma unroll
                for (int nt = 0; nt < 4; ++nt)
                    ph[row * PS + nt * 16 + col_l] = (h16)(v0[nt] * inv);
            }
        }
    }

    // ---- stage 3: o_h = p @ v_h  (per-head; own-wave LDS ordering) ----
    {
        const h16* ph = pls + h * NPAD * PS;
        v8f oacc[4][2];
#pragma unroll
        for (int mt = 0; mt < 4; ++mt)
#pragma unroll
            for (int nt = 0; nt < 2; ++nt) {
                v8f c = 0.0f;
#pragma unroll
                for (int kk = 0; kk < 2; ++kk) {
                    v16h a = frag_A(ph + mt * 16 * PS + kk * 32, PS, lane);
                    v16h b = frag_B(vls + (h * DH + nt * 16) * VS + kk * 32, VS, lane);
                    c = wmma16(a, b, c);
                }
                oacc[mt][nt] = c;
            }
#pragma unroll
        for (int mt = 0; mt < 4; ++mt)
#pragma unroll
            for (int nt = 0; nt < 2; ++nt)
#pragma unroll
                for (int r = 0; r < 8; ++r)
                    oA[(mt * 16 + rofs + r) * XS + h * DH + nt * 16 + col_l] =
                        (h16)oacc[mt][nt][r];
    }
    __syncthreads();

    // ---- stage 4: out = o @ proj_w + proj_b ; 4 x 16 tiles ----
    for (int t = wave; t < 64; t += 8) {
        int mt = t & 3, nt = t >> 2;
        if (t + 8 < 64)
            __builtin_prefetch(pt + ((t + 8) >> 2) * 16 * 256, 0, 3);
        v8f c = 0.0f;
#pragma unroll
        for (int kk = 0; kk < 8; ++kk) {
            v16h a = frag_A(oA + mt * 16 * XS + kk * 32, XS, lane);
            v16h b = frag_B(pt + (nt * 16) * 256 + kk * 32, 256, lane);
            c = wmma16(a, b, c);
        }
        int col = nt * 16 + col_l;
        float pb = proj_b[col];
#pragma unroll
        for (int r = 0; r < 8; ++r) {
            int row = mt * 16 + rofs + r;
            if (row < NTOK)
                out[((size_t)bid * NTOK + row) * DIMC + col] = c[r] + pb;
        }
    }
}

extern "C" void kernel_launch(void* const* d_in, const int* in_sizes, int n_in,
                              void* d_out, int out_size, void* d_ws, size_t ws_size,
                              hipStream_t stream) {
    const float* x          = (const float*)d_in[0];
    const float* mask       = (const float*)d_in[1];
    const float* qkv_w      = (const float*)d_in[2];
    const float* qkv_b      = (const float*)d_in[3];
    const float* proj_w     = (const float*)d_in[4];
    const float* proj_b     = (const float*)d_in[5];
    const float* bias_table = (const float*)d_in[6];
    const int*   rel_index  = (const int*)d_in[7];
    float* out = (float*)d_out;

    h16*   wt = (h16*)((char*)d_ws + WS_WT);
    h16*   pt = (h16*)((char*)d_ws + WS_PT);
    float* bm = (float*)((char*)d_ws + WS_BM);

    (void)hipFuncSetAttribute((const void*)swin_attn,
                              hipFuncAttributeMaxDynamicSharedMemorySize, LDS_BYTES);

    swin_prep<<<1024, 256, 0, stream>>>(qkv_w, proj_w, bias_table, rel_index, mask,
                                        wt, pt, bm);
    swin_attn<<<4096, 256, LDS_BYTES, stream>>>(x, qkv_b, proj_b, wt, pt, bm, out);

    (void)in_sizes; (void)n_in; (void)out_size; (void)ws_size;
}
